// IndRNNBlock_14181982012196
// MI455X (gfx1250) — matrix operand order repaired and checked by reading, compiled, not run on hardware
//
#include <hip/hip_runtime.h>

typedef __attribute__((ext_vector_type(16))) __bf16        v16bf;
typedef __attribute__((ext_vector_type(8)))  float         v8f;
typedef __attribute__((ext_vector_type(4)))  unsigned int  u32x4;
typedef __attribute__((ext_vector_type(4)))  float         f32x4;

#define TT 1024
#define BB 64
#define DD 512
#define HH 512
#define MM (TT * BB)      /* 65536 rows */
#define KK 512
#define CH (BB * HH)      /* 32768 channels per timestep */

// ---------- bf16 helpers (round to nearest even) ----------
__device__ __forceinline__ unsigned short f2bf(float f) {
  unsigned int u = __builtin_bit_cast(unsigned int, f);
  u += 0x7FFFu + ((u >> 16) & 1u);
  return (unsigned short)(u >> 16);
}
__device__ __forceinline__ unsigned int pack2(float lo, float hi) {
  return (unsigned int)f2bf(lo) | ((unsigned int)f2bf(hi) << 16);
}

union Frag {
  v16bf v;
  u32x4 q[2];
};

// ---------------------------------------------------------------------
// Bulk fp32 -> bf16 conversion (memory bound, 8 elems/thread).
// ---------------------------------------------------------------------
__global__ void __launch_bounds__(256)
cvt_kernel(const float* __restrict__ x, unsigned short* __restrict__ xb) {
  const size_t i = ((size_t)blockIdx.x * 256 + threadIdx.x) * 8;
  const f32x4* s = reinterpret_cast<const f32x4*>(x + i);
  f32x4 v0 = s[0], v1 = s[1];
  u32x4 o;
  o[0] = pack2(v0[0], v0[1]); o[1] = pack2(v0[2], v0[3]);
  o[2] = pack2(v1[0], v1[1]); o[3] = pack2(v1[2], v1[3]);
  *reinterpret_cast<u32x4*>(xb + i) = o;
}

// ---------------------------------------------------------------------
// Prep: Wc = w0 @ fc_w  (fp32 accumulate -> bf16), bc = w0 @ fc_b + b0,
//       W1b = bf16(w1).  grid = 512 (h rows), block = 128.
// ---------------------------------------------------------------------
__global__ void __launch_bounds__(128)
prep_kernel(const float* __restrict__ fc_w, const float* __restrict__ fc_b,
            const float* __restrict__ w0,   const float* __restrict__ b0,
            const float* __restrict__ w1,
            unsigned short* __restrict__ Wc, unsigned short* __restrict__ W1b,
            float* __restrict__ bc) {
  const int h   = blockIdx.x;
  const int tid = threadIdx.x;              // 0..127
  const float* w0r = w0 + (size_t)h * DD;
  float a0 = 0.f, a1 = 0.f, a2 = 0.f, a3 = 0.f;
  for (int e = 0; e < DD; ++e) {
    const float w = w0r[e];
    const float* fr = fc_w + (size_t)e * DD + tid;
    a0 = fmaf(w, fr[0],   a0);
    a1 = fmaf(w, fr[128], a1);
    a2 = fmaf(w, fr[256], a2);
    a3 = fmaf(w, fr[384], a3);
  }
  Wc[(size_t)h * DD + tid      ] = f2bf(a0);
  Wc[(size_t)h * DD + tid + 128] = f2bf(a1);
  Wc[(size_t)h * DD + tid + 256] = f2bf(a2);
  Wc[(size_t)h * DD + tid + 384] = f2bf(a3);

  const float* w1r = w1 + (size_t)h * HH;
  W1b[(size_t)h * HH + tid      ] = f2bf(w1r[tid      ]);
  W1b[(size_t)h * HH + tid + 128] = f2bf(w1r[tid + 128]);
  W1b[(size_t)h * HH + tid + 256] = f2bf(w1r[tid + 256]);
  W1b[(size_t)h * HH + tid + 384] = f2bf(w1r[tid + 384]);

  if (tid == 0) {
    float s = 0.f;
    for (int e = 0; e < DD; ++e) s = fmaf(w0r[e], fc_b[e], s);
    bc[h] = s + b0[h];
  }
}

// ---------------------------------------------------------------------
// GEMM: out[M,512] = A[M,512] @ W[512,512]^T + bias   (bf16 WMMA, f32 acc)
//   block tile 128x128, 8 waves (4 along M x 2 along N), wave tile 32x64
//   = 2x4 WMMA 16x16 accumulators, 8 v_wmma per k-step per wave.
//   A (bf16) double-buffered through LDS (row stride 112 B: conflict-free
//   b128 fragment reads); B fragments straight from global (W row-major ==
//   B column-major, matching the documented B VGPR layout; L2 resident).
//   Single barrier per k-step (double buffer removes the WAR sync).
// ---------------------------------------------------------------------
__global__ void __launch_bounds__(256)
gemm_kernel(const unsigned short* __restrict__ A,
            const unsigned short* __restrict__ W,
            const float* __restrict__ bias, float* __restrict__ out) {
  __shared__ __align__(16) unsigned short As[2][128][56];

  const int tid  = threadIdx.x;
  const int lane = tid & 31;
  const int wave = tid >> 5;     // 0..7
  const int wm   = wave & 3;     // 4 waves along M (32 rows each)
  const int wn   = wave >> 2;    // 2 waves along N (64 cols each)
  const int m0   = blockIdx.x * 128;
  const int n0   = blockIdx.y * 128;
  const int ln   = lane & 15;
  const int lh   = lane >> 4;    // 0|1: lane half

  v8f acc[2][4] = {};

  const int srow = tid >> 1;           // 0..127
  const int scol = (tid & 1) * 16;     // 0|16
  const unsigned short* aptr = A + (size_t)(m0 + srow) * KK + scol;

  for (int kb = 0; kb < KK / 32; ++kb) {
    const int k0 = kb * 32;
    // ---- stage A tile (128x32 bf16) into LDS buffer kb&1 ----
    const u32x4* sv = reinterpret_cast<const u32x4*>(aptr + k0);
    u32x4 s0 = sv[0], s1 = sv[1];
    unsigned short* d = &As[kb & 1][srow][scol];
    *reinterpret_cast<u32x4*>(d)     = s0;
    *reinterpret_cast<u32x4*>(d + 8) = s1;
    __syncthreads();

    Frag a[2], b[4];
#pragma unroll
    for (int tm = 0; tm < 2; ++tm) {
      // A layout: lanes 0-15 -> K 0..7 / 16..23, lanes 16-31 -> K 8..15 / 24..31
      const unsigned short* p = &As[kb & 1][wm * 32 + tm * 16 + ln][lh * 8];
      a[tm].q[0] = *reinterpret_cast<const u32x4*>(p);
      a[tm].q[1] = *reinterpret_cast<const u32x4*>(p + 16);
    }
#pragma unroll
    for (int tn = 0; tn < 4; ++tn) {
      // B layout: lanes 0-15 -> K 0..7 / 16..23, lanes 16-31 -> K 8..15 / 24..31
      const unsigned short* p =
          W + (size_t)(n0 + wn * 64 + tn * 16 + ln) * KK + k0 + lh * 8;
      b[tn].q[0] = *reinterpret_cast<const u32x4*>(p);
      b[tn].q[1] = *reinterpret_cast<const u32x4*>(p + 16);
    }
#pragma unroll
    for (int tm = 0; tm < 2; ++tm)
#pragma unroll
      for (int tn = 0; tn < 4; ++tn)
        acc[tm][tn] = __builtin_amdgcn_wmma_f32_16x16x32_bf16(
            false, a[tm].v, false, b[tn].v, (short)0, acc[tm][tn], false, false);
  }

#pragma unroll
  for (int tm = 0; tm < 2; ++tm) {
#pragma unroll
    for (int tn = 0; tn < 4; ++tn) {
      const int n  = n0 + wn * 64 + tn * 16 + ln;
      const float bv = bias[n];
#pragma unroll
      for (int i = 0; i < 8; ++i) {
        const int m = m0 + wm * 32 + tm * 16 + lh * 8 + i;  // C layout: VGPR i -> M
        out[(size_t)m * HH + n] = acc[tm][tn][i] + bv;
      }
    }
  }
}

// ---------------------------------------------------------------------
// IndRNN scans: 32768 independent channels, one thread each, stream T.
// ---------------------------------------------------------------------
__global__ void __launch_bounds__(256)
scan0_kernel(const float* __restrict__ pre, const float* __restrict__ hidden,
             const float* __restrict__ u, unsigned short* __restrict__ out0,
             float* __restrict__ dout) {
  const int idx = blockIdx.x * 256 + threadIdx.x;   // 0..32767
  const int h   = idx & (HH - 1);
  float hh = hidden[idx];
  const float uu = u[h];
#pragma unroll 4
  for (int t = 0; t < TT; ++t) {
    const float p = pre[(size_t)t * CH + idx];
    hh = fmaxf(fmaf(uu, hh, p), 0.0f);
    out0[(size_t)t * CH + idx] = f2bf(hh);          // bf16: next GEMM operand
  }
  dout[(size_t)TT * CH + idx] = hh;                 // hn[0]
}

__global__ void __launch_bounds__(256)
scan1_kernel(const float* __restrict__ pre, const float* __restrict__ hidden,
             const float* __restrict__ u, float* __restrict__ dout) {
  const int idx = blockIdx.x * 256 + threadIdx.x;
  const int h   = idx & (HH - 1);
  float hh = hidden[CH + idx];
  const float uu = u[h];
#pragma unroll 4
  for (int t = 0; t < TT; ++t) {
    const float p = pre[(size_t)t * CH + idx];
    hh = fmaxf(fmaf(uu, hh, p), 0.0f);
    dout[(size_t)t * CH + idx] = hh;                // out1 (fp32 output)
  }
  dout[(size_t)TT * CH + CH + idx] = hh;            // hn[1]
}

// ---------------------------------------------------------------------
extern "C" void kernel_launch(void* const* d_in, const int* in_sizes, int n_in,
                              void* d_out, int out_size, void* d_ws, size_t ws_size,
                              hipStream_t stream) {
  const float* x    = (const float*)d_in[0];
  const float* hid  = (const float*)d_in[1];
  const float* fc_w = (const float*)d_in[2];
  const float* fc_b = (const float*)d_in[3];
  const float* w0   = (const float*)d_in[4];
  const float* b0   = (const float*)d_in[5];
  const float* u0   = (const float*)d_in[6];
  const float* w1   = (const float*)d_in[7];
  const float* b1   = (const float*)d_in[8];
  const float* u1   = (const float*)d_in[9];
  float* out = (float*)d_out;

  char* ws = (char*)d_ws;
  float*          pre   = (float*)ws;                                  // 128 MB, reused
  unsigned short* shbuf = (unsigned short*)(ws + (size_t)134217728);   // 64 MB: xb, then out0
  unsigned short* Wc    = (unsigned short*)(ws + (size_t)201326592);   // 512 KB
  unsigned short* W1b   = (unsigned short*)(ws + (size_t)201850880);   // 512 KB
  float*          bc    = (float*)(ws + (size_t)202375168);            //   2 KB

  // 1) x -> bf16 once (kills redundant fp32 reads + in-loop conversion)
  cvt_kernel<<<(MM * KK) / (256 * 8), 256, 0, stream>>>(x, shbuf);
  // 2) fuse fc into layer-0 projection: Wc = w0@fc_w, bc = w0@fc_b + b0
  prep_kernel<<<512, 128, 0, stream>>>(fc_w, fc_b, w0, b0, w1, Wc, W1b, bc);
  // 3) pre0 = x @ Wc^T + bc      (xb consumed before scan0 reuses the buffer)
  gemm_kernel<<<dim3(MM / 128, HH / 128), 256, 0, stream>>>(shbuf, Wc, bc, pre);
  // 4) layer-0 recurrence -> out0 (bf16, aliases xb) + hn[0]
  scan0_kernel<<<CH / 256, 256, 0, stream>>>(pre, hid, u0, shbuf, out);
  // 5) pre1 = out0 @ w1^T + b1
  gemm_kernel<<<dim3(MM / 128, HH / 128), 256, 0, stream>>>(shbuf, W1b, b1, pre);
  // 6) layer-1 recurrence -> out1 + hn[1]
  scan1_kernel<<<CH / 256, 256, 0, stream>>>(pre, hid, u1, out);
}
